// LinearBlock_26079041421438
// MI455X (gfx1250) — 8-bit matrix operand order repaired and checked by reading, compiled, not run on hardware
//
#include <hip/hip_runtime.h>
#include <stdint.h>

typedef int v8i __attribute__((ext_vector_type(8)));

constexpr int Bm   = 16384;
constexpr int INn  = 2048;
constexpr int OUTn = 2048;
constexpr int KT = INn / 64;
constexpr int MT = Bm / 16;
constexpr int NT = OUTn / 16;

constexpr size_t APACK_BYTES = (size_t)MT * KT * 1024;
constexpr size_t BPACK_BYTES = (size_t)NT * KT * 1024;

__global__ __launch_bounds__(256) void quant_pack_x(const float* __restrict__ x,
                                                    uint8_t* __restrict__ apack) {
  unsigned d    = blockIdx.x * blockDim.x + threadIdx.x;
  unsigned v    = d & 7u;
  unsigned lane = (d >> 3) & 31u;
  unsigned kt   = (d >> 8) & (unsigned)(KT - 1);
  unsigned mt   = d >> 13;
  unsigned half = lane >> 4;
  unsigned m    = mt * 16u + (lane & 15u);
  unsigned kb   = ((v >> 1) * 16u) + ((v & 1u) * 4u) + half * 8u;

  const float* src = x + (size_t)m * INn + (size_t)kt * 64u + kb;
  unsigned out = 0u;
#pragma unroll
  for (int b = 0; b < 4; ++b) {
    float xc   = fminf(fmaxf(src[b], 0.0f), 1.0f);
    unsigned q = (unsigned)(int)rintf(xc * 255.0f);
    out |= (q & 0xffu) << (8 * b);
  }
  ((unsigned*)apack)[d] = out;
}

__global__ __launch_bounds__(256) void quant_pack_w(const float* __restrict__ w,
                                                    uint8_t* __restrict__ bpack) {
  unsigned d    = blockIdx.x * blockDim.x + threadIdx.x;
  unsigned v    = d & 7u;
  unsigned lane = (d >> 3) & 31u;
  unsigned kt   = (d >> 8) & (unsigned)(KT - 1);
  unsigned nt   = d >> 13;
  unsigned half = lane >> 4;
  unsigned oc   = nt * 16u + (lane & 15u);
  unsigned kb   = ((v >> 1) * 16u) + ((v & 1u) * 4u) + half * 8u;

  unsigned p = oc >> 3, o = oc & 7u;
  unsigned out = 0u;
#pragma unroll
  for (int b = 0; b < 4; ++b) {
    unsigned ic = kt * 64u + kb + (unsigned)b;
    unsigned q = ic >> 3, i = ic & 7u;
    size_t flat = ((size_t)(p * 256u + q)) * 64u + o * 8u + i;
    float wc = fminf(fmaxf(w[flat], -2.0f), 2.0f);
    int wi   = (int)rintf(wc * 63.5f);
    out |= ((unsigned)(wi & 0xff)) << (8 * b);
  }
  ((unsigned*)bpack)[d] = out;
}

__global__ __launch_bounds__(256) void gemm_iu8(const uint8_t* __restrict__ apack,
                                                const uint8_t* __restrict__ bpack,
                                                float* __restrict__ out) {
  const int lane = threadIdx.x & 31;
  const int wave = threadIdx.x >> 5;
  const int wm = wave >> 1;
  const int wn = wave & 1;
  const int mtb = blockIdx.y * 8 + wm * 2;
  const int ntb = blockIdx.x * 8 + wn * 4;

  v8i acc[2][4];
#pragma unroll
  for (int i = 0; i < 2; ++i)
#pragma unroll
    for (int j = 0; j < 4; ++j) acc[i][j] = (v8i)(0);

  const size_t laneOff = (size_t)lane * 32u;

  for (int kt = 0; kt < KT; ++kt) {
    v8i a[2], b[4];
#pragma unroll
    for (int i = 0; i < 2; ++i) {
      const uint8_t* ap = apack + ((size_t)(mtb + i) * KT + kt) * 1024u + laneOff;
      a[i] = *(const v8i*)ap;
    }
#pragma unroll
    for (int j = 0; j < 4; ++j) {
      const uint8_t* bp = bpack + ((size_t)(ntb + j) * KT + kt) * 1024u + laneOff;
      b[j] = *(const v8i*)bp;
    }
#pragma unroll
    for (int i = 0; i < 2; ++i)
#pragma unroll
      for (int j = 0; j < 4; ++j)
        acc[i][j] = __builtin_amdgcn_wmma_i32_16x16x64_iu8(
            false, a[i], true, b[j], acc[i][j], false, false);
  }

  const float S = 2.0f / (255.0f * 127.0f);
  const int nlo  = lane & 15;
  const int mrow = (lane >> 4) * 8;

#pragma unroll
  for (int i = 0; i < 2; ++i) {
    const int mbase = (mtb + i) * 16 + mrow;
#pragma unroll
    for (int j = 0; j < 4; ++j) {
      const int n = (ntb + j) * 16 + nlo;
#pragma unroll
      for (int r = 0; r < 8; ++r) {
        float f = (float)acc[i][j][r] * S;
        f = f * f;
        f = fminf(fmaxf(f, 0.0f), 6.0f);
        out[(size_t)(mbase + r) * OUTn + n] = f;
      }
    }
  }
}

extern "C" void kernel_launch(void* const* d_in, const int* in_sizes, int n_in,
                              void* d_out, int out_size, void* d_ws, size_t ws_size,
                              hipStream_t stream) {
  const float* x = (const float*)d_in[0];
  const float* w = (const float*)d_in[1];
  float* out     = (float*)d_out;

  uint8_t* apack = (uint8_t*)d_ws;
  uint8_t* bpack = (uint8_t*)d_ws + APACK_BYTES;
  (void)in_sizes; (void)n_in; (void)out_size; (void)ws_size;

  quant_pack_x<<<(unsigned)(APACK_BYTES / 4 / 256), 256, 0, stream>>>(x, apack);
  quant_pack_w<<<(unsigned)(BPACK_BYTES / 4 / 256), 256, 0, stream>>>(w, bpack);
  dim3 grid(OUTn / 128, Bm / 128);
  gemm_iu8<<<grid, 256, 0, stream>>>(apack, bpack, out);
}
